// Decoder_26061861552325
// MI455X (gfx1250) — matrix operand order repaired and checked by reading, not compiled, not run on hardware
//
#include <hip/hip_runtime.h>
#include <hip/hip_bf16.h>
#include <stdint.h>

// Problem dims (fixed by the reference)
#define BB   32
#define SS   400
#define TT   100
#define VV   32000
#define OOVV 100
#define VEXT (VV + OOVV)
#define EE   512
#define UU   256
#define AUU  256
#define GPP  256
#define ENCC 512

typedef __bf16 bf16_t;
typedef __attribute__((ext_vector_type(16))) __bf16 v16bf;
typedef __attribute__((ext_vector_type(8)))  __bf16 v8bf;
typedef __attribute__((ext_vector_type(8)))  float  v8f;

__device__ __forceinline__ bf16_t f2bf(float f) {
  unsigned u = __float_as_uint(f);
  unsigned r = (u + 0x7FFFu + ((u >> 16) & 1u)) >> 16;   // RNE
  unsigned short s = (unsigned short)r;
  return __builtin_bit_cast(bf16_t, s);
}
__device__ __forceinline__ float sigmoidf_(float x) { return 1.0f / (1.0f + __expf(-x)); }

// ---------------------------------------------------------------------------
// Weight re-pack: fp32 row-major [K,N] -> bf16 WMMA B-fragment tiles.
// CDNA5 16-bit B (32x16): lane 0-15 -> elem 0..7 = K0..7, 8..15 = K16..23;
// lane 16-31 -> K8..15 / K24..31; N = lane&15. Tile = 512 bf16, lane-major (lane*16 + j).
// Tile order: nt-major, kt inner (streaming K for a fixed N tile).
// ---------------------------------------------------------------------------
__global__ void k_pack_b(const float* __restrict__ W, int N, int Ktiles,
                         bf16_t* __restrict__ out) {
  int nt = blockIdx.x, kt = blockIdx.y, tid = threadIdx.x;
  bf16_t* dst = out + ((size_t)nt * Ktiles + kt) * 512;
  for (int e = tid; e < 512; e += 256) {
    int lane = e >> 4, j = e & 15;
    int k   = kt * 32 + ((lane >> 4) << 3) + (j & 7) + ((j >> 3) << 4);
    int col = nt * 16 + (lane & 15);
    dst[e] = f2bf(W[(size_t)k * N + col]);
  }
}

// Teacher-forced inputs: xs[t][b][e] = bf16(emb[gt_tokens[b][t]][e])
__global__ void k_gather_emb(const int* __restrict__ gt, const float* __restrict__ emb,
                             bf16_t* __restrict__ xs) {
  int idx = blockIdx.x * blockDim.x + threadIdx.x;
  if (idx >= TT * BB * EE) return;
  int e = idx & (EE - 1);
  int tb = idx >> 9;          // /512
  int b = tb & (BB - 1);
  int t = tb >> 5;            // /32
  int tok = gt[b * TT + t];
  xs[idx] = f2bf(emb[(size_t)tok * EE + e]);
}

__global__ void k_init(const float* __restrict__ h0, const float* __restrict__ c0,
                       float* __restrict__ h, float* __restrict__ c,
                       bf16_t* __restrict__ hbf, float* __restrict__ cov) {
  int idx = blockIdx.x * blockDim.x + threadIdx.x;
  if (idx < BB * UU) { float hv = h0[idx]; h[idx] = hv; c[idx] = c0[idx]; hbf[idx] = f2bf(hv); }
  if (idx < BB * SS) cov[idx] = 0.0f;
}

__global__ void k_f2bf(const float* __restrict__ in, bf16_t* __restrict__ out, int n) {
  int i = blockIdx.x * blockDim.x + threadIdx.x;
  if (i < n) out[i] = f2bf(in[i]);
}

// ---------------------------------------------------------------------------
// A-fragment load, CDNA5 16-bit A (16x32) layout: lane<16 -> row=lane,
// v[0..7]=K0..7, v[8..15]=K16..23 ; lane>=16 -> row=lane-16, K8..15 / K24..31.
// ---------------------------------------------------------------------------
__device__ __forceinline__ v16bf load_a_frag(const bf16_t* __restrict__ A, int lda, int lane) {
  int row = lane & 15;
  int khalf = (lane >> 4) << 3;
  const bf16_t* base = A + (size_t)row * lda;
  v8bf lo = *(const v8bf*)(base + khalf);        // 16B load
  v8bf hi = *(const v8bf*)(base + 16 + khalf);   // 16B load
  v16bf v;
#pragma unroll
  for (int j = 0; j < 8; ++j) { v[j] = lo[j]; v[8 + j] = hi[j]; }
  return v;
}

// Generic C = A[32,K] * W[K,16*gridDim.x] (+bias) (+=C if accumulate).
// Block = 64 threads = 2 waves = the two M tiles of B=32. Grid.x = N tiles.
__global__ void k_wmma_gemm(const bf16_t* __restrict__ A, int lda,
                            const bf16_t* __restrict__ Wp, int Ktiles,
                            const float* __restrict__ bias,
                            float* __restrict__ C, int ldc, int accumulate) {
  int lane = threadIdx.x & 31;
  int mt   = threadIdx.x >> 5;          // wave id: M tile 0 or 1
  int nt   = blockIdx.x;
  const bf16_t* Arow = A + (size_t)(mt * 16) * lda;
  const bf16_t* Wt   = Wp + (size_t)nt * Ktiles * 512 + (size_t)lane * 16;
  v8f acc = {};
  for (int kt = 0; kt < Ktiles; ++kt) {
    v16bf a = load_a_frag(Arow + kt * 32, lda, lane);
    v16bf b = *(const v16bf*)(Wt + (size_t)kt * 512);   // 32B contiguous/lane
    acc = __builtin_amdgcn_wmma_f32_16x16x32_bf16(false, a, false, b,
                                                  (short)0, acc, false, false);
  }
  // C/D layout: elem r -> M = mt*16 + (lane>>4)*8 + r, N = lane&15
  int col   = nt * 16 + (lane & 15);
  int rbase = mt * 16 + ((lane >> 4) << 3);
  float bv = bias ? bias[col] : 0.0f;
#pragma unroll
  for (int r = 0; r < 8; ++r) {
    size_t idx = (size_t)(rbase + r) * ldc + col;
    float prev = accumulate ? C[idx] : 0.0f;
    C[idx] = prev + acc[r] + bv;
  }
}

// LSTM gates (TF order i,f,g,o); also emits bf16 h and bf16 dec_state=[h,c].
__global__ void k_lstm_gates(const float* __restrict__ z, float* __restrict__ h,
                             float* __restrict__ c, bf16_t* __restrict__ hbf,
                             bf16_t* __restrict__ dsbf) {
  int idx = blockIdx.x * blockDim.x + threadIdx.x;
  if (idx >= BB * UU) return;
  int b = idx >> 8, u = idx & 255;
  const float* zr = z + (size_t)b * (4 * UU);
  float i_ = zr[u], f_ = zr[UU + u], g_ = zr[2 * UU + u], o_ = zr[3 * UU + u];
  float cn = sigmoidf_(f_) * c[idx] + sigmoidf_(i_) * tanhf(g_);
  float hn = sigmoidf_(o_) * tanhf(cn);
  h[idx] = hn; c[idx] = cn;
  hbf[idx] = f2bf(hn);
  dsbf[(size_t)b * (2 * UU) + u]      = f2bf(hn);
  dsbf[(size_t)b * (2 * UU) + UU + u] = f2bf(cn);
}

// Bahdanau attention with coverage; one block per batch row.
__global__ void k_attention(const float* __restrict__ enc_attn,
                            const float* __restrict__ enc_out,
                            const float* __restrict__ dec_attn,
                            const float* __restrict__ Wc, const float* __restrict__ Wa,
                            float* __restrict__ cov, float* __restrict__ a_out,
                            float* __restrict__ cvec, float* __restrict__ covloss) {
  __shared__ float sc[SS];
  __shared__ float red[256];
  int b = blockIdx.x, tid = threadIdx.x;
  const float* da   = dec_attn + (size_t)b * AUU;
  const float* covr = cov + (size_t)b * SS;
  for (int s = tid; s < SS; s += 256) {
    const float* ea = enc_attn + ((size_t)b * SS + s) * AUU;
    float cv = covr[s], acc = 0.0f;
    for (int k = 0; k < AUU; ++k)
      acc += tanhf(ea[k] + cv * Wc[k] + da[k]) * Wa[k];
    sc[s] = acc;
  }
  __syncthreads();
  float m = -INFINITY;
  for (int s = tid; s < SS; s += 256) m = fmaxf(m, sc[s]);
  red[tid] = m; __syncthreads();
  for (int o = 128; o > 0; o >>= 1) { if (tid < o) red[tid] = fmaxf(red[tid], red[tid + o]); __syncthreads(); }
  float mx = red[0]; __syncthreads();
  float ssum = 0.0f;
  for (int s = tid; s < SS; s += 256) { float e = __expf(sc[s] - mx); sc[s] = e; ssum += e; }
  red[tid] = ssum; __syncthreads();
  for (int o = 128; o > 0; o >>= 1) { if (tid < o) red[tid] += red[tid + o]; __syncthreads(); }
  float inv = 1.0f / red[0]; __syncthreads();
  float closs = 0.0f;
  for (int s = tid; s < SS; s += 256) {
    float av = sc[s] * inv;
    sc[s] = av;
    a_out[(size_t)b * SS + s] = av;
    closs += fminf(covr[s], av);          // uses cov BEFORE update
    cov[(size_t)b * SS + s] = covr[s] + av;
  }
  red[tid] = closs; __syncthreads();
  for (int o = 128; o > 0; o >>= 1) { if (tid < o) red[tid] += red[tid + o]; __syncthreads(); }
  if (tid == 0) covloss[(size_t)b * TT] = red[0];
  __syncthreads();
  for (int e = tid; e < ENCC; e += 256) {
    float acc = 0.0f;
    const float* eo = enc_out + (size_t)b * SS * ENCC + e;
    for (int s = 0; s < SS; ++s) acc += sc[s] * eo[(size_t)s * ENCC];
    cvec[(size_t)b * ENCC + e] = acc;
  }
}

// pg scalar quirk: uses only batch row 0 of dec_state, c_vec and x.
__global__ void k_pg(const float* __restrict__ h, const float* __restrict__ c,
                     const float* __restrict__ cvec, const int* __restrict__ gt,
                     const float* __restrict__ emb,
                     const float* __restrict__ Ws, const float* __restrict__ Wcp,
                     const float* __restrict__ Wi, const float* __restrict__ bi,
                     const float* __restrict__ Wg, int t, float* __restrict__ pg_out) {
  __shared__ float red[256];
  int p = threadIdx.x;
  float g = bi[p];
  for (int k = 0; k < UU; ++k)   g += h[k]    * Ws[(size_t)k * GPP + p];
  for (int k = 0; k < UU; ++k)   g += c[k]    * Ws[(size_t)(UU + k) * GPP + p];
  for (int k = 0; k < ENCC; ++k) g += cvec[k] * Wcp[(size_t)k * GPP + p];
  const float* x0 = emb + (size_t)gt[t] * EE;   // gt_tokens[0][t]
  for (int k = 0; k < EE; ++k)   g += x0[k]   * Wi[(size_t)k * GPP + p];
  red[p] = g * Wg[p];
  __syncthreads();
  for (int o = 128; o > 0; o >>= 1) { if (p < o) red[p] += red[p + o]; __syncthreads(); }
  if (p == 0) *pg_out = 1.0f / (1.0f + __expf(-red[0]));
}

// ---------------------------------------------------------------------------
// Extended softmax + copy-dist scatter + argmax, staged entirely in LDS
// (125.4 KB row buffer -- CDNA5 allows up to 320 KB per workgroup, needs
// dynamic LDS). Final row streamed to d_out with NON-TEMPORAL stores so the
// 411 MB write-once probs stream doesn't evict the L2-resident weights, and
// nothing is ever read back from global.
// ---------------------------------------------------------------------------
__global__ void k_final(const float* __restrict__ logits, const float* __restrict__ a,
                        const int* __restrict__ ext, const float* __restrict__ pg_p,
                        float* __restrict__ probs_out, float* __restrict__ seq_out) {
  extern __shared__ float row[];        // VEXT floats
  __shared__ float red[256];
  __shared__ int   redi[256];
  int b = blockIdx.x, tid = threadIdx.x;
  const float* lg = logits + (size_t)b * VV;
  float* out = probs_out + (size_t)b * TT * VEXT;   // row (b, t) of d_out
  float pg = *pg_p;
  float m = -INFINITY;
  for (int j = tid; j < VV; j += 256) m = fmaxf(m, lg[j]);
  red[tid] = m; __syncthreads();
  for (int o = 128; o > 0; o >>= 1) { if (tid < o) red[tid] = fmaxf(red[tid], red[tid + o]); __syncthreads(); }
  float mx = red[0]; __syncthreads();
  float ssum = 0.0f;
  for (int j = tid; j < VV; j += 256) ssum += __expf(lg[j] - mx);
  red[tid] = ssum; __syncthreads();
  for (int o = 128; o > 0; o >>= 1) { if (tid < o) red[tid] += red[tid + o]; __syncthreads(); }
  float scale = pg / red[0]; __syncthreads();
  for (int j = tid; j < VV; j += 256) row[j] = __expf(lg[j] - mx) * scale;
  for (int j = VV + tid; j < VEXT; j += 256) row[j] = 0.0f;
  __syncthreads();
  if (tid == 0) {                        // deterministic scatter of (1-pg)*a in LDS
    float w = 1.0f - pg;
    const int* er = ext + (size_t)b * SS;
    const float* ar = a + (size_t)b * SS;
    for (int s = 0; s < SS; ++s) row[er[s]] += w * ar[s];
  }
  __syncthreads();
  // argmax (first-occurrence tie-break) out of LDS
  float bv = -INFINITY; int bi_ = 0;
  for (int j = tid; j < VEXT; j += 256) { float v = row[j]; if (v > bv) { bv = v; bi_ = j; } }
  red[tid] = bv; redi[tid] = bi_; __syncthreads();
  for (int o = 128; o > 0; o >>= 1) {
    if (tid < o) {
      if (red[tid + o] > red[tid] ||
          (red[tid + o] == red[tid] && redi[tid + o] < redi[tid])) {
        red[tid] = red[tid + o]; redi[tid] = redi[tid + o];
      }
    }
    __syncthreads();
  }
  if (tid == 0) ((int*)seq_out)[(size_t)b * TT] = redi[0];  // int32 bits
  // stream the finished row out with non-temporal stores
  for (int j = tid; j < VEXT; j += 256)
    __builtin_nontemporal_store(row[j], &out[j]);
}

static inline size_t align256(size_t x) { return (x + 255) & ~(size_t)255; }

extern "C" void kernel_launch(void* const* d_in, const int* in_sizes, int n_in,
                              void* d_out, int out_size, void* d_ws, size_t ws_size,
                              hipStream_t stream) {
  (void)in_sizes; (void)n_in; (void)out_size; (void)ws_size;
  const int*   gt         = (const int*)d_in[0];
  const int*   ext        = (const int*)d_in[1];
  const float* enc_output = (const float*)d_in[2];
  const float* enc_attn   = (const float*)d_in[3];
  const float* h0         = (const float*)d_in[4];
  const float* c0         = (const float*)d_in[5];
  const float* emb        = (const float*)d_in[6];
  const float* lstm_kernel= (const float*)d_in[7];
  const float* lstm_rec   = (const float*)d_in[8];
  const float* lstm_bias  = (const float*)d_in[9];
  const float* attn_Wd    = (const float*)d_in[10];
  const float* attn_bd    = (const float*)d_in[11];
  const float* attn_Wc    = (const float*)d_in[12];
  const float* attn_Wa    = (const float*)d_in[13];
  const float* clf_W1     = (const float*)d_in[14];
  const float* clf_b1     = (const float*)d_in[15];
  const float* clf_W2     = (const float*)d_in[16];
  const float* clf_b2     = (const float*)d_in[17];
  const float* gp_Ws      = (const float*)d_in[18];
  const float* gp_Wc      = (const float*)d_in[19];
  const float* gp_Wi      = (const float*)d_in[20];
  const float* gp_bi      = (const float*)d_in[21];
  const float* gp_Wg      = (const float*)d_in[22];

  uint8_t* ws = (uint8_t*)d_ws;
  size_t off = 0;
  auto alloc = [&](size_t bytes) -> void* { void* p = ws + off; off = align256(off + bytes); return p; };

  // bf16 fragment-packed weights (~18 MB) + activations (~8 MB): clf_W2 at
  // 16 MB bf16 stays resident in the 192 MB L2 across all 100 steps.
  bf16_t* WKp  = (bf16_t*)alloc((size_t)64 * 16 * 512 * 2);   // lstm_kernel 512x1024
  bf16_t* WRp  = (bf16_t*)alloc((size_t)64 * 8  * 512 * 2);   // lstm_rec    256x1024
  bf16_t* WDp  = (bf16_t*)alloc((size_t)16 * 16 * 512 * 2);   // attn_Wd     512x256
  bf16_t* W1p  = (bf16_t*)alloc((size_t)16 * 8  * 512 * 2);   // clf_W1      256x256
  bf16_t* W2p  = (bf16_t*)alloc((size_t)2000 * 8 * 512 * 2);  // clf_W2      256x32000
  bf16_t* XS   = (bf16_t*)alloc((size_t)TT * BB * EE * 2);    // xs [T,B,E]
  bf16_t* HBF  = (bf16_t*)alloc((size_t)BB * UU * 2);
  bf16_t* H1BF = (bf16_t*)alloc((size_t)BB * UU * 2);
  bf16_t* DSBF = (bf16_t*)alloc((size_t)BB * 2 * UU * 2);
  float*  H    = (float*)alloc((size_t)BB * UU * 4);
  float*  Cc   = (float*)alloc((size_t)BB * UU * 4);
  float*  COV  = (float*)alloc((size_t)BB * SS * 4);
  float*  Aw   = (float*)alloc((size_t)BB * SS * 4);
  float*  CVEC = (float*)alloc((size_t)BB * ENCC * 4);
  float*  DATT = (float*)alloc((size_t)BB * AUU * 4);
  float*  Z    = (float*)alloc((size_t)BB * 4 * UU * 4);
  float*  H1   = (float*)alloc((size_t)BB * UU * 4);
  float*  LOG  = (float*)alloc((size_t)BB * VV * 4);
  float*  PG   = (float*)alloc(4);

  dim3 b256(256);
  // One-time prep (re-run every call: deterministic, no cached state).
  k_pack_b<<<dim3(64, 16),  b256, 0, stream>>>(lstm_kernel, 4 * UU, 16, WKp);
  k_pack_b<<<dim3(64, 8),   b256, 0, stream>>>(lstm_rec,    4 * UU, 8,  WRp);
  k_pack_b<<<dim3(16, 16),  b256, 0, stream>>>(attn_Wd,     AUU,    16, WDp);
  k_pack_b<<<dim3(16, 8),   b256, 0, stream>>>(clf_W1,      UU,     8,  W1p);
  k_pack_b<<<dim3(2000, 8), b256, 0, stream>>>(clf_W2,      VV,     8,  W2p);
  k_gather_emb<<<(TT * BB * EE + 255) / 256, b256, 0, stream>>>(gt, emb, XS);
  k_init<<<(BB * SS + 255) / 256, b256, 0, stream>>>(h0, c0, H, Cc, HBF, COV);

  float* out = (float*)d_out;
  size_t seqbase = (size_t)BB * TT * VEXT;
  size_t covbase = seqbase + (size_t)BB * TT;
  size_t smem_final = (size_t)VEXT * sizeof(float);   // 125.4 KB dynamic LDS

  for (int t = 0; t < TT; ++t) {
    // z = x_t @ Wk + b ; z += h_{t-1} @ Wr
    k_wmma_gemm<<<64, 64, 0, stream>>>(XS + (size_t)t * BB * EE, EE, WKp, 16, lstm_bias, Z, 4 * UU, 0);
    k_wmma_gemm<<<64, 64, 0, stream>>>(HBF, UU, WRp, 8, nullptr, Z, 4 * UU, 1);
    k_lstm_gates<<<(BB * UU + 255) / 256, b256, 0, stream>>>(Z, H, Cc, HBF, DSBF);
    // dec_attn = [h,c] @ attn_Wd + bd
    k_wmma_gemm<<<16, 64, 0, stream>>>(DSBF, 2 * UU, WDp, 16, attn_bd, DATT, AUU, 0);
    k_attention<<<BB, b256, 0, stream>>>(enc_attn, enc_output, DATT, attn_Wc, attn_Wa,
                                         COV, Aw, CVEC, out + covbase + t);
    k_pg<<<1, b256, 0, stream>>>(H, Cc, CVEC, gt, emb, gp_Ws, gp_Wc, gp_Wi, gp_bi, gp_Wg, t, PG);
    // h1 = h @ clf_W1 + b1 ; logits = h1 @ clf_W2 + b2
    k_wmma_gemm<<<16, 64, 0, stream>>>(HBF, UU, W1p, 8, clf_b1, H1, UU, 0);
    k_f2bf<<<(BB * UU + 255) / 256, b256, 0, stream>>>(H1, H1BF, BB * UU);
    k_wmma_gemm<<<2000, 64, 0, stream>>>(H1BF, UU, W2p, 8, clf_b2, LOG, VV, 0);
    k_final<<<BB, b256, smem_final, stream>>>(LOG, Aw, ext, PG, out + (size_t)t * VEXT, out + seqbase + t);
  }
}
